// SGCNLayer_14809047236866
// MI455X (gfx1250) — compile-verified, matrix operand order checked by reading, not run on hardware
//
#include <hip/hip_runtime.h>
#include <math.h>

typedef __attribute__((ext_vector_type(16))) __bf16 v16bf;
typedef __attribute__((ext_vector_type(8)))  __bf16 v8bf;
typedef __attribute__((ext_vector_type(8)))  float  v8f;
typedef __attribute__((ext_vector_type(4)))  float  v4f;

#define B_ 4
#define L_ 4
#define N_ 2048
#define H_ 64
#define V_ 4

// ---------------------------------------------------------------------------
// Kernel 1: h = tanh(LN(concat(x,feat) @ fc_w^T + fc_b))   [B,N,64] fp32
// one 64-thread block per (b,n) row; LDS tree reduction for mean/var
// ---------------------------------------------------------------------------
__global__ __launch_bounds__(64) void sgcn_fc_ln_tanh(
    const float* __restrict__ x, const float* __restrict__ feat,
    const float* __restrict__ fc_w, const float* __restrict__ fc_b,
    const float* __restrict__ ln_g, const float* __restrict__ ln_b,
    float* __restrict__ h)
{
    const int bn = blockIdx.x;        // 0 .. B*N-1
    const int j  = threadIdx.x;       // output channel 0..63
    const float* xr = x    + (size_t)bn * H_;
    const float* fr = feat + (size_t)bn * 2;
    const float* wr = fc_w + (size_t)j * (H_ + 2);
    float s = fc_b[j];
    #pragma unroll 8
    for (int k = 0; k < H_; ++k) s += xr[k] * wr[k];
    s += fr[0] * wr[H_] + fr[1] * wr[H_ + 1];

    __shared__ float s1[64];
    __shared__ float s2[64];
    s1[j] = s; s2[j] = s * s;
    __syncthreads();
    for (int off = 32; off > 0; off >>= 1) {
        if (j < off) { s1[j] += s1[j + off]; s2[j] += s2[j + off]; }
        __syncthreads();
    }
    const float mu  = s1[0] * (1.0f / H_);
    const float var = s2[0] * (1.0f / H_) - mu * mu;
    const float hn  = (s - mu) * rsqrtf(var + 1e-5f) * ln_g[j] + ln_b[j];
    h[(size_t)bn * H_ + j] = tanhf(hn);
}

// ---------------------------------------------------------------------------
// Kernel 2: M[l,d,e] = sum_v a[l,v] * W[d,v,e]   [5,64,64] fp32  (tiny)
// ---------------------------------------------------------------------------
__global__ __launch_bounds__(256) void sgcn_make_M(
    const float* __restrict__ a, const float* __restrict__ W,
    float* __restrict__ M)
{
    const int l = blockIdx.x;         // 0..4
    for (int i = threadIdx.x; i < H_ * H_; i += 256) {
        const int d = i >> 6, e = i & 63;
        float s = 0.0f;
        #pragma unroll
        for (int v = 0; v < V_; ++v)
            s += a[l * V_ + v] * W[((size_t)d * V_ + v) * H_ + e];
        M[(size_t)l * (H_ * H_) + i] = s;
    }
}

// ---------------------------------------------------------------------------
// Kernel 3: fold per-label matrices into h.
//   base[b,n,e] = 0.2*(h@M[0])[b,n,e] + x[b,n,e]          fp32
//   hMT[b,l,e,n] = bf16( (h@M[l+1])[b,n,e] )              (transposed: K=n contiguous)
// ---------------------------------------------------------------------------
__global__ __launch_bounds__(64) void sgcn_fold(
    const float* __restrict__ h, const float* __restrict__ M,
    const float* __restrict__ x, float* __restrict__ base,
    __bf16* __restrict__ hMT)
{
    const int bn = blockIdx.x;
    const int b  = bn >> 11;
    const int n  = bn & (N_ - 1);
    const int e  = threadIdx.x;
    __shared__ float hr[64];
    hr[e] = h[(size_t)bn * H_ + e];
    __syncthreads();

    float s0 = 0.0f;
    #pragma unroll 8
    for (int d = 0; d < H_; ++d) s0 += hr[d] * M[d * H_ + e];
    base[(size_t)bn * H_ + e] = 0.2f * s0 + x[(size_t)bn * H_ + e];

    for (int l = 0; l < L_; ++l) {
        const float* Ml = M + (size_t)(l + 1) * (H_ * H_);
        float s = 0.0f;
        #pragma unroll 8
        for (int d = 0; d < H_; ++d) s += hr[d] * Ml[d * H_ + e];
        hMT[(((size_t)b * L_ + l) * H_ + e) * N_ + n] = (__bf16)s;
    }
}

// ---------------------------------------------------------------------------
// Kernel 4: out[b] = base[b] + 0.2 * sum_l adj[b,l] @ hM[b,l]
// bf16 WMMA 16x16x32; 8 waves/block, each wave = 16 rows x 64 cols of C.
// adj is a 256 MB single-use stream -> non-temporal loads (keep L2 for hMT).
// out is write-once -> non-temporal stores.
// ---------------------------------------------------------------------------
__device__ inline v8f wmma_step(v16bf a, const __bf16* __restrict__ bp, v8f c)
{
    v8bf b0 = *(const v8bf*)(bp);
    v8bf b1 = *(const v8bf*)(bp + 16);
    v16bf bf;
    #pragma unroll
    for (int i = 0; i < 8; ++i) { bf[i] = b0[i]; bf[8 + i] = b1[i]; }
    return __builtin_amdgcn_wmma_f32_16x16x32_bf16(
        /*neg_a=*/false, a, /*neg_b=*/false, bf,
        /*c_mod=*/(short)0, c, /*reuse_a=*/false, /*reuse_b=*/false);
}

__global__ __launch_bounds__(256) void sgcn_bmm_wmma(
    const float* __restrict__ adj, const __bf16* __restrict__ hMT,
    const float* __restrict__ base, float* __restrict__ out)
{
    const int b    = blockIdx.y;
    const int wave = threadIdx.x >> 5;
    const int lane = threadIdx.x & 31;
    const int r    = lane & 15;
    const int kh   = lane >> 4;                  // K-half selector (0/1)
    const int row0 = blockIdx.x * 128 + wave * 16;
    const int n    = row0 + r;                   // A-matrix row for this lane

    v8f acc0 = {}, acc1 = {}, acc2 = {}, acc3 = {};

    for (int l = 0; l < L_; ++l) {
        const float*  arow  = adj + (((size_t)b * L_ + l) * N_ + n) * N_;
        const __bf16* bbase = hMT + ((size_t)b * L_ + l) * (size_t)H_ * N_;
        for (int m0 = 0; m0 < N_; m0 += 32) {
            // ---- A fragment: 16 fp32 adj values -> bf16, ISA layout ----
            // single-use stream: non-temporal so it doesn't evict hMT from L2
            const float* ap = arow + m0 + kh * 8;
            v4f f0 = __builtin_nontemporal_load((const v4f*)(ap));
            v4f f1 = __builtin_nontemporal_load((const v4f*)(ap + 4));
            v4f f2 = __builtin_nontemporal_load((const v4f*)(ap + 16));
            v4f f3 = __builtin_nontemporal_load((const v4f*)(ap + 20));
            v16bf afrag;
            #pragma unroll
            for (int i = 0; i < 4; ++i) {
                afrag[i]      = (__bf16)f0[i];
                afrag[4 + i]  = (__bf16)f1[i];
                afrag[8 + i]  = (__bf16)f2[i];
                afrag[12 + i] = (__bf16)f3[i];
            }
            // ---- B fragments (4 e-tiles, cache-resident) + WMMA ----
            const __bf16* bp = bbase + (size_t)r * N_ + m0 + kh * 8;
            acc0 = wmma_step(afrag, bp,                     acc0);
            acc1 = wmma_step(afrag, bp + (size_t)16 * N_,   acc1);
            acc2 = wmma_step(afrag, bp + (size_t)32 * N_,   acc2);
            acc3 = wmma_step(afrag, bp + (size_t)48 * N_,   acc3);
        }
    }

    // ---- epilogue: C layout, VGPR j -> row j + kh*8, col r ----
    const int orow0 = row0 + kh * 8;
    #pragma unroll
    for (int j = 0; j < 8; ++j) {
        const size_t o = ((size_t)b * N_ + orow0 + j) * H_ + r;
        __builtin_nontemporal_store(0.2f * acc0[j] + base[o],      out + o);
        __builtin_nontemporal_store(0.2f * acc1[j] + base[o + 16], out + o + 16);
        __builtin_nontemporal_store(0.2f * acc2[j] + base[o + 32], out + o + 32);
        __builtin_nontemporal_store(0.2f * acc3[j] + base[o + 48], out + o + 48);
    }
}

// ---------------------------------------------------------------------------
extern "C" void kernel_launch(void* const* d_in, const int* in_sizes, int n_in,
                              void* d_out, int out_size, void* d_ws, size_t ws_size,
                              hipStream_t stream)
{
    (void)in_sizes; (void)n_in; (void)out_size; (void)ws_size;
    const float* x    = (const float*)d_in[0];
    const float* feat = (const float*)d_in[1];
    const float* adj  = (const float*)d_in[2];
    const float* fc_w = (const float*)d_in[3];
    const float* fc_b = (const float*)d_in[4];
    const float* ln_g = (const float*)d_in[5];
    const float* ln_b = (const float*)d_in[6];
    const float* W    = (const float*)d_in[7];
    const float* a    = (const float*)d_in[8];
    float* out = (float*)d_out;

    char* ws = (char*)d_ws;
    float*  h    = (float*)(ws);                    // 4*2048*64*4   = 2 MB
    float*  M    = (float*)(ws + 2097152);          // 5*64*64*4     = 80 KB
    float*  base = (float*)(ws + 2179072);          // 4*2048*64*4   = 2 MB
    __bf16* hMT  = (__bf16*)(ws + 4276224);         // 4*4*64*2048*2 = 4 MB

    sgcn_fc_ln_tanh<<<B_ * N_, 64, 0, stream>>>(x, feat, fc_w, fc_b, ln_g, ln_b, h);
    sgcn_make_M<<<L_ + 1, 256, 0, stream>>>(a, W, M);
    sgcn_fold<<<B_ * N_, 64, 0, stream>>>(h, M, x, base, hMT);
    dim3 grid(N_ / 128, B_);
    sgcn_bmm_wmma<<<grid, 256, 0, stream>>>(adj, hMT, base, out);
}
